// WNN_62629213110312
// MI455X (gfx1250) — matrix operand order repaired and checked by reading, not compiled, not run on hardware
//
#include <hip/hip_runtime.h>

typedef __attribute__((ext_vector_type(16))) _Float16 v16h;
typedef __attribute__((ext_vector_type(8)))  _Float16 v8h;
typedef __attribute__((ext_vector_type(8)))  float    v8f;

#define LP 136  // padded LDS pitch (halfs) for 128-wide rows: 272B row stride -> conflict-free 16-row b128 access

#define WMMA_F16(a, b, c) \
  __builtin_amdgcn_wmma_f32_16x16x32_f16(false, (a), false, (b), (short)0, (c), false, false)

// A operand (16x32 MxK, f16): lane<16 -> row=lane, K {0..7, 16..23}; lane>=16 -> row=lane-16, K {8..15, 24..31}
__device__ __forceinline__ v16h frag_a(const _Float16* p, int pitch, int row, int kb, int lane) {
  const _Float16* q = p + (row + (lane & 15)) * pitch + kb + ((lane >> 4) << 3);
  union { v16h v; v8h h[2]; } u;
  u.h[0] = *(const v8h*)q;
  u.h[1] = *(const v8h*)(q + 16);
  return u.v;
}

// B operand (32x16 KxN, f16) from an [n][k] store: lane holds col n=lane&15, K {0..7, 16..23} (lane<16) / {8..15, 24..31} (lane>=16)
__device__ __forceinline__ v16h frag_b(const _Float16* p, int pitch, int col, int kb, int lane) {
  const _Float16* q = p + (col + (lane & 15)) * pitch + kb + ((lane >> 4) << 3);
  union { v16h v; v8h h[2]; } u;
  u.h[0] = *(const v8h*)q;
  u.h[1] = *(const v8h*)(q + 16);
  return u.v;
}

__global__ __launch_bounds__(256, 1)
void jacnet_kernel(const float* __restrict__ u,
                   const float* __restrict__ W1, const float* __restrict__ b1,
                   const float* __restrict__ W2, const float* __restrict__ b2,
                   const float* __restrict__ W3, const float* __restrict__ b3,
                   const float* __restrict__ W4,
                   float* __restrict__ out) {
  __shared__ alignas(16) _Float16 sW2[128 * LP];   // W2 f16
  __shared__ alignas(16) _Float16 sW3[128 * LP];   // W3 f16
  __shared__ alignas(16) _Float16 sR1[128 * LP];   // a1 [s][k]  ->  JT_a [n][k]
  __shared__ alignas(16) _Float16 sR2[128 * LP];   // a2 [s][k]  ->  JT_b [n][k] -> Jf f32 scratch
  __shared__ alignas(16) _Float16 sW4p[16 * 128];  // W4 padded to 16 rows, [n][k]
  __shared__ float sW1f[512];                      // W1 f32 [128][4]
  __shared__ float sB1[128], sB2[128], sB3[128];
  __shared__ unsigned sM1[512], sM2[512], sM3[512]; // ReLU masks, 128 bits/sample

  const int tid  = threadIdx.x;
  const int lane = tid & 31;
  const int wv   = tid >> 5;   // 8 waves of 32

  // ---- stage weights into LDS (f32 -> f16), zero mask accumulators ----
  for (int i = tid; i < 128 * 128; i += 256) {
    const int r = i >> 7, c = i & 127;
    sW2[r * LP + c] = (_Float16)W2[i];
    sW3[r * LP + c] = (_Float16)W3[i];
  }
  for (int i = tid; i < 2048; i += 256)
    sW4p[i] = (i < 512) ? (_Float16)W4[i] : (_Float16)0.0f;
  for (int i = tid; i < 512; i += 256) {
    sW1f[i] = W1[i];
    sM2[i] = 0u;
    sM3[i] = 0u;
  }
  if (tid < 128) { sB1[tid] = b1[tid]; sB2[tid] = b2[tid]; sB3[tid] = b3[tid]; }
  __syncthreads();

  // ---- layer 1 (K=4): VALU. a1 = relu(u@W1^T + b1), mask1 bits ----
  {
    const int sl = tid >> 1;               // local sample 0..127
    const int jb = (tid & 1) << 6;         // hidden half 0 / 64
    const float4 uu = ((const float4*)u)[(size_t)blockIdx.x * 128 + sl];
    unsigned bits0 = 0u, bits1 = 0u;
    for (int j = 0; j < 64; ++j) {
      const int jj = jb + j;
      float h = sB1[jj] + uu.x * sW1f[jj * 4 + 0] + uu.y * sW1f[jj * 4 + 1]
                        + uu.z * sW1f[jj * 4 + 2] + uu.w * sW1f[jj * 4 + 3];
      const bool m = h > 0.0f;
      sR1[sl * LP + jj] = (_Float16)(m ? h : 0.0f);
      if (j < 32) bits0 |= (unsigned)m << j; else bits1 |= (unsigned)m << (j - 32);
    }
    sM1[sl * 4 + (jb >> 5) + 0] = bits0;
    sM1[sl * 4 + (jb >> 5) + 1] = bits1;
  }
  __syncthreads();

  // ---- layer 2: h2^T = W2 @ a1^T via WMMA; a2 = relu(h2+b2) -> sR2, mask2 ----
  for (int st = 0; st < 8; ++st) {
    v8f acc = {};
#pragma unroll
    for (int kk = 0; kk < 128; kk += 32)
      acc = WMMA_F16(frag_a(sW2, LP, wv * 16, kk, lane),
                     frag_b(sR1, LP, st * 16, kk, lane), acc);
    const int s  = st * 16 + (lane & 15);
    const int i0 = wv * 16 + ((lane >> 4) << 3);
    unsigned bits = 0u;
    v8h ov;
#pragma unroll
    for (int r = 0; r < 8; ++r) {
      const float h = acc[r] + sB2[i0 + r];
      const bool m = h > 0.0f;
      bits |= (unsigned)m << r;
      ov[r] = (_Float16)(m ? h : 0.0f);
    }
    *(v8h*)&sR2[s * LP + i0] = ov;
    atomicOr(&sM2[s * 4 + (i0 >> 5)], bits << (i0 & 31));
  }
  __syncthreads();

  // ---- layer 3: only need mask3 = (W3 @ a2^T + b3 > 0) ----
  for (int st = 0; st < 8; ++st) {
    v8f acc = {};
#pragma unroll
    for (int kk = 0; kk < 128; kk += 32)
      acc = WMMA_F16(frag_a(sW3, LP, wv * 16, kk, lane),
                     frag_b(sR2, LP, st * 16, kk, lane), acc);
    const int s  = st * 16 + (lane & 15);
    const int i0 = wv * 16 + ((lane >> 4) << 3);
    unsigned bits = 0u;
#pragma unroll
    for (int r = 0; r < 8; ++r)
      bits |= (unsigned)((acc[r] + sB3[i0 + r]) > 0.0f) << r;
    atomicOr(&sM3[s * 4 + (i0 >> 5)], bits << (i0 & 31));
  }

  // ---- Jacobian chain: 4 chunks of 32 samples (128 stacked columns each) ----
  float* sD = (float*)sR2;  // Jf scratch reuses region 2 after jac L3
  for (int c = 0; c < 4; ++c) {
    const int sb = c * 32;  // local sample base of chunk
    __syncthreads();

    // J0^T [n=(s,k4)][j] = m1[s][j] ? W1[j][k4] : 0  -> sR1
    {
      const int n  = tid >> 1;
      const int jb = (tid & 1) << 6;
      const int s  = sb + (n >> 2);
      const int k4 = n & 3;
      for (int j = 0; j < 64; ++j) {
        const int jj = jb + j;
        const bool m = (sM1[s * 4 + (jj >> 5)] >> (jj & 31)) & 1u;
        sR1[n * LP + jj] = m ? (_Float16)sW1f[jj * 4 + k4] : (_Float16)0.0f;
      }
    }
    __syncthreads();

    // J2 = m2 .* (W2 @ J0):  A=W2, B=JT_a -> JT_b
    for (int nt = 0; nt < 8; ++nt) {
      v8f acc = {};
#pragma unroll
      for (int kk = 0; kk < 128; kk += 32)
        acc = WMMA_F16(frag_a(sW2, LP, wv * 16, kk, lane),
                       frag_b(sR1, LP, nt * 16, kk, lane), acc);
      const int n  = nt * 16 + (lane & 15);
      const int s  = sb + (n >> 2);
      const int m0 = wv * 16 + ((lane >> 4) << 3);
      const unsigned bits = (sM2[s * 4 + (m0 >> 5)] >> (m0 & 31)) & 0xffu;
      v8h ov;
#pragma unroll
      for (int r = 0; r < 8; ++r)
        ov[r] = ((bits >> r) & 1u) ? (_Float16)acc[r] : (_Float16)0.0f;
      *(v8h*)&sR2[n * LP + m0] = ov;
    }
    __syncthreads();

    // J3 = m3 .* (W3 @ J2):  A=W3, B=JT_b -> JT_a
    for (int nt = 0; nt < 8; ++nt) {
      v8f acc = {};
#pragma unroll
      for (int kk = 0; kk < 128; kk += 32)
        acc = WMMA_F16(frag_a(sW3, LP, wv * 16, kk, lane),
                       frag_b(sR2, LP, nt * 16, kk, lane), acc);
      const int n  = nt * 16 + (lane & 15);
      const int s  = sb + (n >> 2);
      const int m0 = wv * 16 + ((lane >> 4) << 3);
      const unsigned bits = (sM3[s * 4 + (m0 >> 5)] >> (m0 & 31)) & 0xffu;
      v8h ov;
#pragma unroll
      for (int r = 0; r < 8; ++r)
        ov[r] = ((bits >> r) & 1u) ? (_Float16)acc[r] : (_Float16)0.0f;
      *(v8h*)&sR1[n * LP + m0] = ov;
    }
    __syncthreads();

    // Jf^T: D[m=(s,k4)][i] = sum_j JT_a[m][j] * W4[i][j]   (A=JT_a, B=W4 padded to N=16)
    {
      v8f acc = {};
#pragma unroll
      for (int kk = 0; kk < 128; kk += 32)
        acc = WMMA_F16(frag_a(sR1, LP, wv * 16, kk, lane),
                       frag_b(sW4p, 128, 0, kk, lane), acc);
      const int n  = lane & 15;
      const int m0 = wv * 16 + ((lane >> 4) << 3);
#pragma unroll
      for (int r = 0; r < 8; ++r)
        sD[(m0 + r) * 16 + n] = acc[r];
    }
    __syncthreads();

    // out[s][x][y] = Jf[s][y][x] - Jf[s][x][y] = D[(s,x)][y] - D[(s,y)][x]
    for (int t = tid; t < 512; t += 256) {
      const int sl = t >> 4, x = (t >> 2) & 3, y = t & 3;
      const float val = sD[(sl * 4 + x) * 16 + y] - sD[(sl * 4 + y) * 16 + x];
      out[((size_t)blockIdx.x * 128 + sb + sl) * 16 + x * 4 + y] = val;
    }
  }
}

extern "C" void kernel_launch(void* const* d_in, const int* in_sizes, int n_in,
                              void* d_out, int out_size, void* d_ws, size_t ws_size,
                              hipStream_t stream) {
  const float* u  = (const float*)d_in[0];
  const float* W1 = (const float*)d_in[1];
  const float* b1 = (const float*)d_in[2];
  const float* W2 = (const float*)d_in[3];
  const float* b2 = (const float*)d_in[4];
  const float* W3 = (const float*)d_in[5];
  const float* b3 = (const float*)d_in[6];
  const float* W4 = (const float*)d_in[7];
  float* out = (float*)d_out;

  const int B = in_sizes[0] / 4;      // IN_DIM = 4
  const int blocks = B / 128;         // 128 samples per block (B = 131072 -> 1024 blocks)
  jacnet_kernel<<<dim3(blocks), dim3(256), 0, stream>>>(u, W1, b1, W2, b2, W3, b3, W4, out);

  (void)n_in; (void)out_size; (void)d_ws; (void)ws_size;
}
